// Sem_Seg_86672440033836
// MI455X (gfx1250) — matrix operand order repaired and checked by reading, not compiled, not run on hardware
//
#include <hip/hip_runtime.h>
#include <hip/hip_bf16.h>

typedef __attribute__((ext_vector_type(16))) _Float16 v16h;
typedef __attribute__((ext_vector_type(8)))  _Float16 v8h;
typedef __attribute__((ext_vector_type(8)))  float    v8f;
typedef __attribute__((ext_vector_type(2)))  float    v2f;

#define KNN 30

// ---------------- WMMA wrappers (CDNA5 gfx1250) ----------------
static __device__ inline v8f wmma_f32x4(v2f a, v2f b, v8f c) {
  // D = A(16x4 f32) * B(4x16 f32) + C, exact f32 semantics for kNN distances
  return __builtin_amdgcn_wmma_f32_16x16x4_f32(false, a, false, b, (short)0, c, false, false);
}
static __device__ inline v8f wmma_f16x32(v16h a, v16h b, v8f c) {
  return __builtin_amdgcn_wmma_f32_16x16x32_f16(false, a, false, b, (short)0, c, false, false);
}

// A fragment (16x32 f16): lane m owns row m; contiguous k order matching B:
// [k0+16*hi, +8) -> elements 0..7, [k0+16*hi+8, +8) -> 8..15.
// Two 16-byte vector loads instead of 16 scalar u16 loads.
static __device__ inline v16h load_arow(const _Float16* ar, int hi, int k0) {
  const v8h c0 = *(const v8h*)(ar + k0 + 16 * hi);
  const v8h c1 = *(const v8h*)(ar + k0 + 16 * hi + 8);
  v16h a;
#pragma unroll
  for (int e = 0; e < 8; ++e) { a[e] = c0[e]; a[8 + e] = c1[e]; }
  return a;
}
// B fragment from pre-shuffled weights Ws[kt][col][32] (t = 16*hi + e):
// one contiguous 32-byte read per lane.
static __device__ inline v16h load_bshuf(const _Float16* Ws, int Nn, int kt, int col, int hi) {
  return *(const v16h*)(Ws + ((size_t)(kt * Nn + col) << 5) + hi * 16);
}

// ---------------- prep kernels ----------------
__global__ void fold_bn(const float* __restrict__ g, const float* __restrict__ be,
                        const float* __restrict__ mn, const float* __restrict__ vr,
                        float* __restrict__ a, float* __restrict__ c, int n) {
  int i = blockIdx.x * 256 + threadIdx.x;
  if (i < n) {
    float av = g[i] * rsqrtf(vr[i] + 1e-5f);
    a[i] = av;
    c[i] = be[i] - mn[i] * av;
  }
}
// Convert f32 W[K x Nn] -> f16 fragment-shuffled layout [Kpad/32][Nn][32],
// dst[(kt*Nn+col)*32 + t] = W[(kt*32+t)*Nn + col], zero-padded K rows.
__global__ void wshuf(const float* __restrict__ src, _Float16* __restrict__ dst,
                      int K, int Nn, int Kpad) {
  int i = blockIdx.x * 256 + threadIdx.x;
  if (i < Kpad * Nn) {
    int t = i & 31;
    int rest = i >> 5;
    int col = rest % Nn;
    int kt = rest / Nn;
    int k = kt * 32 + t;
    dst[i] = (_Float16)(k < K ? src[(size_t)k * Nn + col] : 0.0f);
  }
}
__global__ void padx(const float* __restrict__ x, float* __restrict__ fp, int N) {
  int i = blockIdx.x * 256 + threadIdx.x;
  if (i < N) {
    fp[i * 4 + 0] = x[i * 3 + 0];
    fp[i * 4 + 1] = x[i * 3 + 1];
    fp[i * 4 + 2] = x[i * 3 + 2];
    fp[i * 4 + 3] = 0.0f;
  }
}
__global__ void sqnorm(const float* __restrict__ f, int ld, int d,
                       float* __restrict__ sq, int N) {
  int i = blockIdx.x * 256 + threadIdx.x;
  if (i < N) {
    float s = 0.0f;
    for (int t = 0; t < d; ++t) { float v = f[(size_t)i * ld + t]; s += v * v; }
    sq[i] = s;
  }
}
__global__ void convact(const float* __restrict__ src, _Float16* __restrict__ dst,
                        unsigned long long n) {
  unsigned long long i = (unsigned long long)blockIdx.x * 256u + threadIdx.x;
  if (i < n) dst[i] = (_Float16)src[i];
}

// ---------------- kNN: WMMA f32 Gram tiles + streaming top-30 ----------------
template <int NKT>
__global__ __launch_bounds__(32) void knn_kernel(const float* __restrict__ f, int ld,
                                                 const float* __restrict__ sq,
                                                 int* __restrict__ idx, int N) {
  __shared__ float sdist[16][17];
  __shared__ float best_d[16][KNN];
  __shared__ int   best_i[16][KNN];
  const int lane = threadIdx.x;
  const int m = lane & 15, hi = lane >> 4;
  const int row0 = blockIdx.x * 16;

  v2f afrag[NKT];
#pragma unroll
  for (int kt = 0; kt < NKT; ++kt) {
    const float* ap = f + (size_t)(row0 + m) * ld + kt * 4 + 2 * hi;
    afrag[kt].x = ap[0];
    afrag[kt].y = ap[1];
  }
  float sqrow[8];
#pragma unroll
  for (int v = 0; v < 8; ++v) sqrow[v] = sq[row0 + v + 8 * hi];

  float worst = 3.0e38f;
  int wslot = 0;
  if (lane < 16)
    for (int j = 0; j < KNN; ++j) { best_d[lane][j] = 3.0e38f; best_i[lane][j] = 0; }
  __syncthreads();

  for (int ct = 0; ct < N / 16; ++ct) {
    const int col0 = ct * 16;
    v8f acc = {};
#pragma unroll
    for (int kt = 0; kt < NKT; ++kt) {
      v2f bfrag;
      const float* bp = f + (size_t)(col0 + m) * ld + kt * 4 + 2 * hi;
      bfrag.x = bp[0];
      bfrag.y = bp[1];
      acc = wmma_f32x4(afrag[kt], bfrag, acc);
    }
    float sqc = sq[col0 + m];  // this lane owns column n = m
#pragma unroll
    for (int v = 0; v < 8; ++v)
      sdist[v + 8 * hi][m] = sqrow[v] + sqc - 2.0f * acc[v];
    __syncthreads();
    if (lane < 16) {
      const int r = lane;
      for (int c = 0; c < 16; ++c) {
        float dv = sdist[r][c];
        if (dv < worst) {  // threshold-reject; accepted ~k*ln(N/k) times
          best_d[r][wslot] = dv;
          best_i[r][wslot] = col0 + c;
          worst = -3.0e38f;
          for (int j = 0; j < KNN; ++j) {
            float bj = best_d[r][j];
            if (bj > worst) { worst = bj; wslot = j; }
          }
        }
      }
    }
    __syncthreads();
  }
  if (lane < 16)
    for (int j = 0; j < KNN; ++j)
      idx[(size_t)(row0 + lane) * KNN + j] = best_i[lane][j];
}

// ---------------- EdgeConv MLP: gather -> (2d->64->64) -> max over k ----------
// One wave owns 16 points; W1/W2 (fragment-shuffled) staged in LDS once and
// reused across all 30 neighbor slots.
template <int D, int KPAD>
__global__ __launch_bounds__(32) void edge_kernel(
    const float* __restrict__ f, int ldf, const int* __restrict__ idx,
    const _Float16* __restrict__ W1, const float* __restrict__ b1,
    const float* __restrict__ a1, const float* __restrict__ c1,
    const _Float16* __restrict__ W2, const float* __restrict__ b2,
    const float* __restrict__ a2, const float* __restrict__ c2,
    float* __restrict__ out, int ldo) {
  __shared__ alignas(32) _Float16 eA[16 * KPAD];
  __shared__ alignas(32) _Float16 h1[16 * 64];
  __shared__ alignas(32) _Float16 sW1[KPAD * 64];
  __shared__ alignas(32) _Float16 sW2[64 * 64];
  const int lane = threadIdx.x;
  const int m = lane & 15, hi = lane >> 4;
  const int p0 = blockIdx.x * 16;

  // stage shuffled weights into LDS (16-byte chunks per lane)
  for (int i = lane; i < (KPAD * 64) / 8; i += 32)
    ((v8h*)sW1)[i] = ((const v8h*)W1)[i];
  for (int i = lane; i < (64 * 64) / 8; i += 32)
    ((v8h*)sW2)[i] = ((const v8h*)W2)[i];

  v8f mx[4];
  for (int nt = 0; nt < 4; ++nt)
#pragma unroll
    for (int v = 0; v < 8; ++v) mx[nt][v] = -3.0e38f;
  __syncthreads();

  for (int j = 0; j < KNN; ++j) {
    if (lane < 16) {  // build edge features e = [xi, xj-xi], zero-padded to KPAD
      const int pi = p0 + lane;
      const int nj = idx[(size_t)pi * KNN + j];
      const float* xi = f + (size_t)pi * ldf;
      const float* xj = f + (size_t)nj * ldf;
      _Float16* er = eA + lane * KPAD;
      for (int t = 0; t < D; ++t) {
        float av = xi[t];
        er[t]     = (_Float16)av;
        er[D + t] = (_Float16)(xj[t] - av);
      }
      for (int t = 2 * D; t < KPAD; ++t) er[t] = (_Float16)0.0f;
    }
    __syncthreads();
    // layer 1: [16 x KPAD] @ [KPAD x 64]
    for (int nt = 0; nt < 4; ++nt) {
      v8f c = {};
#pragma unroll
      for (int kt = 0; kt < KPAD / 32; ++kt) {
        v16h a = load_arow(eA + m * KPAD, hi, kt * 32);
        v16h b = load_bshuf(sW1, 64, kt, nt * 16 + m, hi);
        c = wmma_f16x32(a, b, c);
      }
      const int col = nt * 16 + m;
      float bb = b1[col], aa = a1[col], cv = c1[col];
#pragma unroll
      for (int v = 0; v < 8; ++v) {
        float z = c[v] + bb;
        z = z > 0.0f ? z : 0.0f;
        h1[(v + 8 * hi) * 64 + col] = (_Float16)(aa * z + cv);
      }
    }
    __syncthreads();
    // layer 2: [16 x 64] @ [64 x 64], then max-accumulate
    for (int nt = 0; nt < 4; ++nt) {
      v8f c = {};
#pragma unroll
      for (int kt = 0; kt < 2; ++kt) {
        v16h a = load_arow(h1 + m * 64, hi, kt * 32);
        v16h b = load_bshuf(sW2, 64, kt, nt * 16 + m, hi);
        c = wmma_f16x32(a, b, c);
      }
      const int col = nt * 16 + m;
      float bb = b2[col], aa = a2[col], cv = c2[col];
#pragma unroll
      for (int v = 0; v < 8; ++v) {
        float z = c[v] + bb;
        z = z > 0.0f ? z : 0.0f;
        z = aa * z + cv;
        mx[nt][v] = mx[nt][v] > z ? mx[nt][v] : z;
      }
    }
    __syncthreads();
  }
  for (int nt = 0; nt < 4; ++nt)
#pragma unroll
    for (int v = 0; v < 8; ++v)
      out[(size_t)(p0 + v + 8 * hi) * ldo + nt * 16 + m] = mx[nt][v];
}

// ---------------- generic f16 GEMM with fused relu+BN epilogue ----------------
// W is fragment-shuffled; each B fragment is one 32B contiguous read (L2-resident).
__global__ __launch_bounds__(32) void gemm_kernel(
    const _Float16* __restrict__ A, int lda, const _Float16* __restrict__ W, int Nn,
    int K, const float* __restrict__ bias, const float* __restrict__ sa,
    const float* __restrict__ sc, _Float16* __restrict__ out, int ldo) {
  const int lane = threadIdx.x;
  const int m = lane & 15, hi = lane >> 4;
  const int row0 = blockIdx.x * 16;
  const int col0 = blockIdx.y * 64;
  v8f c[4];
  v8f zero = {};
  for (int nt = 0; nt < 4; ++nt) c[nt] = zero;
  const _Float16* ar = A + (size_t)(row0 + m) * lda;
  for (int kt = 0; kt * 32 < K; ++kt) {
    __builtin_prefetch(ar + kt * 32 + 512, 0, 1);  // gfx1250 global_prefetch
    v16h a = load_arow(ar, hi, kt * 32);
    for (int nt = 0; nt < 4; ++nt) {
      v16h b = load_bshuf(W, Nn, kt, col0 + nt * 16 + m, hi);
      c[nt] = wmma_f16x32(a, b, c[nt]);
    }
  }
  for (int nt = 0; nt < 4; ++nt) {
    const int col = col0 + nt * 16 + m;
    float bb = bias[col], aa = sa[col], cv = sc[col];
#pragma unroll
    for (int v = 0; v < 8; ++v) {
      float z = c[nt][v] + bb;
      z = z > 0.0f ? z : 0.0f;
      out[(size_t)(row0 + v + 8 * hi) * ldo + col] = (_Float16)(aa * z + cv);
    }
  }
}

// ---------------- final 128->12 + log_softmax (tiny; VALU) ----------------
__global__ void final_kernel(const _Float16* __restrict__ h, const float* __restrict__ Wf,
                             const float* __restrict__ bf, float* __restrict__ out, int N) {
  int i = blockIdx.x * 128 + threadIdx.x;
  if (i >= N) return;
  float z[12];
  for (int o = 0; o < 12; ++o) z[o] = bf[o];
  for (int k = 0; k < 128; ++k) {
    float hv = (float)h[(size_t)i * 128 + k];
    for (int o = 0; o < 12; ++o) z[o] += hv * Wf[k * 12 + o];
  }
  float mx = z[0];
  for (int o = 1; o < 12; ++o) mx = fmaxf(mx, z[o]);
  float s = 0.0f;
  for (int o = 0; o < 12; ++o) s += expf(z[o] - mx);
  float lse = mx + logf(s);
  for (int o = 0; o < 12; ++o) out[(size_t)i * 12 + o] = z[o] - lse;
}

// ---------------- host orchestration ----------------
// Input flattening assumed in setup_inputs() insertion order:
// 0:x, then per layer {W,b,gamma,beta,mean,var}:
// conv1[0]=1.., conv1[1]=7.., conv2[0]=13.., conv2[1]=19.., conv3[0]=25..,
// conv3[1]=31.., lin1[0]=37.., head[0]=43.., head[1]=49.., final_W=55, final_b=56
extern "C" void kernel_launch(void* const* d_in, const int* in_sizes, int n_in,
                              void* d_out, int out_size, void* d_ws, size_t ws_size,
                              hipStream_t stream) {
  const int N = in_sizes[0] / 3;  // 16384
  const float* x = (const float*)d_in[0];
  auto P = [&](int i) { return (const float*)d_in[i]; };

  char* w = (char*)d_ws;
  size_t off = 0;
  auto alloc = [&](size_t bytes) -> void* {
    void* p = w + off;
    off = (off + bytes + 255) & ~(size_t)255;
    return p;
  };

  float*     fpad4 = (float*)alloc((size_t)N * 4 * 4);
  float*     xcat  = (float*)alloc((size_t)N * 192 * 4);
  float*     sq    = (float*)alloc((size_t)N * 4);
  int*       idx   = (int*)alloc((size_t)N * KNN * 4);
  _Float16*  A0    = (_Float16*)alloc((size_t)N * 192 * 2);
  _Float16*  Hl1   = (_Float16*)alloc((size_t)N * 1024 * 2);
  _Float16*  Hh1   = (_Float16*)alloc((size_t)N * 256 * 2);
  _Float16*  Hh2   = (_Float16*)alloc((size_t)N * 128 * 2);

  const int eK[6]    = {6, 64, 128, 64, 128, 64};
  const int eKp[6]   = {32, 64, 128, 64, 128, 64};
  const int eWidx[6] = {1, 7, 13, 19, 25, 31};
  _Float16* We[6];
  for (int i = 0; i < 6; ++i) We[i] = (_Float16*)alloc((size_t)eKp[i] * 64 * 2);
  _Float16* Wl1 = (_Float16*)alloc((size_t)192 * 1024 * 2);
  _Float16* Wh1 = (_Float16*)alloc((size_t)1024 * 256 * 2);
  _Float16* Wh2 = (_Float16*)alloc((size_t)256 * 128 * 2);

  const int bnN[9]   = {64, 64, 64, 64, 64, 64, 1024, 256, 128};
  const int bnIdx[9] = {1, 7, 13, 19, 25, 31, 37, 43, 49};
  float *sa[9], *sc[9];
  for (int i = 0; i < 9; ++i) {
    sa[i] = (float*)alloc((size_t)bnN[i] * 4);
    sc[i] = (float*)alloc((size_t)bnN[i] * 4);
  }

  // --- prep: fold BN, shuffle+convert weights to f16 fragment layout, pad x ---
  for (int i = 0; i < 9; ++i)
    fold_bn<<<(bnN[i] + 255) / 256, 256, 0, stream>>>(
        P(bnIdx[i] + 2), P(bnIdx[i] + 3), P(bnIdx[i] + 4), P(bnIdx[i] + 5),
        sa[i], sc[i], bnN[i]);
  for (int i = 0; i < 6; ++i) {
    int tot = eKp[i] * 64;
    wshuf<<<(tot + 255) / 256, 256, 0, stream>>>(P(eWidx[i]), We[i], eK[i], 64, eKp[i]);
  }
  wshuf<<<(192 * 1024 + 255) / 256, 256, 0, stream>>>(P(37), Wl1, 192, 1024, 192);
  wshuf<<<(1024 * 256 + 255) / 256, 256, 0, stream>>>(P(43), Wh1, 1024, 256, 1024);
  wshuf<<<(256 * 128 + 255) / 256, 256, 0, stream>>>(P(49), Wh2, 256, 128, 256);
  padx<<<(N + 255) / 256, 256, 0, stream>>>(x, fpad4, N);

  // --- conv1 (d=3 padded to 4) ---
  sqnorm<<<(N + 255) / 256, 256, 0, stream>>>(fpad4, 4, 4, sq, N);
  knn_kernel<1><<<N / 16, 32, 0, stream>>>(fpad4, 4, sq, idx, N);
  edge_kernel<3, 32><<<N / 16, 32, 0, stream>>>(
      fpad4, 4, idx, We[0], P(2), sa[0], sc[0], We[1], P(8), sa[1], sc[1],
      xcat + 0, 192);
  // --- conv2 (d=64) ---
  sqnorm<<<(N + 255) / 256, 256, 0, stream>>>(xcat + 0, 192, 64, sq, N);
  knn_kernel<16><<<N / 16, 32, 0, stream>>>(xcat + 0, 192, sq, idx, N);
  edge_kernel<64, 128><<<N / 16, 32, 0, stream>>>(
      xcat + 0, 192, idx, We[2], P(14), sa[2], sc[2], We[3], P(20), sa[3], sc[3],
      xcat + 64, 192);
  // --- conv3 (d=64) ---
  sqnorm<<<(N + 255) / 256, 256, 0, stream>>>(xcat + 64, 192, 64, sq, N);
  knn_kernel<16><<<N / 16, 32, 0, stream>>>(xcat + 64, 192, sq, idx, N);
  edge_kernel<64, 128><<<N / 16, 32, 0, stream>>>(
      xcat + 64, 192, idx, We[4], P(26), sa[4], sc[4], We[5], P(32), sa[5], sc[5],
      xcat + 128, 192);

  // --- dense MLPs ---
  unsigned long long nact = (unsigned long long)N * 192ull;
  convact<<<(unsigned)((nact + 255) / 256), 256, 0, stream>>>(xcat, A0, nact);
  gemm_kernel<<<dim3(N / 16, 1024 / 64), 32, 0, stream>>>(A0, 192, Wl1, 1024, 192,
                                                          P(38), sa[6], sc[6], Hl1, 1024);
  gemm_kernel<<<dim3(N / 16, 256 / 64), 32, 0, stream>>>(Hl1, 1024, Wh1, 256, 1024,
                                                         P(44), sa[7], sc[7], Hh1, 256);
  gemm_kernel<<<dim3(N / 16, 128 / 64), 32, 0, stream>>>(Hh1, 256, Wh2, 128, 256,
                                                         P(50), sa[8], sc[8], Hh2, 128);
  final_kernel<<<(N + 127) / 128, 128, 0, stream>>>(Hh2, P(55), P(56), (float*)d_out, N);
  (void)n_in; (void)out_size; (void)ws_size;
}
